// GraphWaveNet_31765578121225
// MI455X (gfx1250) — matrix operand order repaired and checked by reading, not compiled, not run on hardware
//
#include <hip/hip_runtime.h>
#include <hip/hip_bf16.h>

typedef unsigned short u16;
typedef unsigned int   u32;
typedef __attribute__((ext_vector_type(16))) __bf16 v16bf;
typedef __attribute__((ext_vector_type(8)))  float  v8f;

union BF16x16 { v16bf v; u32 u[8]; };

__device__ __forceinline__ u16 f2bf(float f) {
    union { float f; u32 u; } c; c.f = f;
    return (u16)(c.u >> 16);
}
__device__ __forceinline__ u32 pack2(float a, float b) {
    return (u32)f2bf(a) | ((u32)f2bf(b) << 16);
}
__device__ __forceinline__ v8f wmma_bf16(const BF16x16& a, const BF16x16& b, v8f c) {
    return __builtin_amdgcn_wmma_f32_16x16x32_bf16(false, a.v, false, b.v, (short)0, c, false, false);
}

#define NN 512   // graph nodes
#define HH 64    // hidden channels
#define TT 64    // time steps
#define BB 16    // batch
#define TOUT 65  // conv output rows per (b,m): t in [0,64]

// ---------------------------------------------------------------------------
// adj = 0.5*static + 0.5*rownorm(relu(e1 @ e2^T)); stored transposed bf16:
// adjT[m*NN + n] = adj[n][m]  (so WMMA-B K(=n) pairs are contiguous)
// ---------------------------------------------------------------------------
__global__ void k_adj(const float* __restrict__ e1, const float* __restrict__ e2,
                      const float* __restrict__ sadj, u16* __restrict__ adjT) {
    __shared__ float row[NN];
    __shared__ float red[256];
    const int n = blockIdx.x, tid = threadIdx.x;
    float a1[16];
    #pragma unroll
    for (int e = 0; e < 16; ++e) a1[e] = e1[n * 16 + e];
    for (int m = tid; m < NN; m += 256) {
        float acc = 0.f;
        #pragma unroll
        for (int e = 0; e < 16; ++e) acc += a1[e] * e2[m * 16 + e];
        row[m] = fmaxf(acc, 0.f);
    }
    __syncthreads();
    red[tid] = row[tid] + row[tid + 256];
    __syncthreads();
    for (int s = 128; s > 0; s >>= 1) {
        if (tid < s) red[tid] += red[tid + s];
        __syncthreads();
    }
    const float inv = 1.f / (red[0] + 1e-8f);
    for (int m = tid; m < NN; m += 256) {
        float a = 0.5f * sadj[n * NN + m] + 0.5f * row[m] * inv;
        adjT[m * NN + n] = f2bf(a);
    }
}

// ---------------------------------------------------------------------------
// start 1x1 conv: h[b][c][t][n] = sw[c][0]*x[b][n][0][t] + sw[c][1]*x[b][n][1][t] + sb[c]
// h stored bf16 with n innermost (contiguous K for diffusion A-tiles)
// ---------------------------------------------------------------------------
__global__ void k_start(const float* __restrict__ x, const float* __restrict__ sw,
                        const float* __restrict__ sb, u16* __restrict__ hbf) {
    const int idx = blockIdx.x * 256 + threadIdx.x; // total = BB*HH*TT*NN
    const int n = idx & (NN - 1);
    const int t = (idx >> 9) & (TT - 1);
    const int c = (idx >> 15) & (HH - 1);
    const int b = idx >> 21;
    const float x0 = x[((b * NN + n) * 2 + 0) * TT + t];
    const float x1 = x[((b * NN + n) * 2 + 1) * TT + t];
    hbf[idx] = f2bf(sw[c * 2] * x0 + sw[c * 2 + 1] * x1 + sb[c]);
}

__global__ void k_zero(float* __restrict__ p, int n) {
    int i = blockIdx.x * 256 + threadIdx.x;
    if (i < n) p[i] = 0.f;
}

// ---------------------------------------------------------------------------
// Graph diffusion GEMM: xs[b][m][t][c] = sum_n h[b][c][t][n] * adjT[m][n]
// M = B*H*T = 65536 rows, K = N = 512, Ncols = 512.  bf16 WMMA, f32 acc.
// 2x2 register tiling: one wave owns a 32x32 output block -> 4 accumulators,
// 4 independent WMMAs per k-step sharing A0/A1/B0/B1 (2x fewer loads/WMMA,
// 2x less L2 re-read of h and adj, and ILP to hide XDL latency).
// ---------------------------------------------------------------------------
__global__ void k_diffuse(const u16* __restrict__ hbf, const u16* __restrict__ adjT,
                          u16* __restrict__ xs) {
    const int gw = blockIdx.x * 4 + (threadIdx.x >> 5);
    const int rowPair = gw >> 4;      // 2048 macro row tiles (32 rows each)
    const int colPair = gw & 15;      // 16 macro col tiles (32 cols each)
    const int lane = threadIdx.x & 31;
    const int mrow = lane & 15;
    const int khalf = lane >> 4;

    const u16* arow0 = hbf + (size_t)(rowPair * 32 + mrow) * NN;
    const u16* arow1 = arow0 + (size_t)16 * NN;
    const u16* brow0 = adjT + (size_t)(colPair * 32 + mrow) * NN; // B col = lane&15
    const u16* brow1 = brow0 + (size_t)16 * NN;

    v8f acc00 = {0.f,0.f,0.f,0.f,0.f,0.f,0.f,0.f};
    v8f acc01 = {0.f,0.f,0.f,0.f,0.f,0.f,0.f,0.f};
    v8f acc10 = {0.f,0.f,0.f,0.f,0.f,0.f,0.f,0.f};
    v8f acc11 = {0.f,0.f,0.f,0.f,0.f,0.f,0.f,0.f};

    for (int kb = 0; kb < NN; kb += 32) {
        BF16x16 A0, A1, B0, B1;
        #pragma unroll
        for (int j = 0; j < 4; ++j) {
            const int ofs = kb + khalf * 8 + 2 * j;
            A0.u[j] = *(const u32*)(arow0 + ofs);
            A1.u[j] = *(const u32*)(arow1 + ofs);
        }
        #pragma unroll
        for (int j = 4; j < 8; ++j) {
            const int ofs = kb + 16 + khalf * 8 + 2 * (j - 4);
            A0.u[j] = *(const u32*)(arow0 + ofs);
            A1.u[j] = *(const u32*)(arow1 + ofs);
        }
        #pragma unroll
        for (int j = 0; j < 8; ++j) {
            const int ofs = kb + (j < 4 ? khalf * 8 + 2 * j : 16 + khalf * 8 + 2 * (j - 4));
            B0.u[j] = *(const u32*)(brow0 + ofs);
            B1.u[j] = *(const u32*)(brow1 + ofs);
        }
        acc00 = wmma_bf16(A0, B0, acc00);
        acc01 = wmma_bf16(A0, B1, acc01);
        acc10 = wmma_bf16(A1, B0, acc10);
        acc11 = wmma_bf16(A1, B1, acc11);
    }

    #pragma unroll
    for (int pi = 0; pi < 2; ++pi) {
        #pragma unroll
        for (int pj = 0; pj < 2; ++pj) {
            const v8f acc = (pi == 0) ? (pj == 0 ? acc00 : acc01)
                                      : (pj == 0 ? acc10 : acc11);
            const int mg = colPair * 32 + pj * 16 + mrow;
            #pragma unroll
            for (int j = 0; j < 8; ++j) {
                const int r2 = rowPair * 32 + pi * 16 + j + 8 * khalf; // (b,c,t)
                const int t = r2 & 63, c = (r2 >> 6) & 63, b = r2 >> 12;
                xs[(((size_t)(b * NN + mg)) * TT + t) * HH + c] = f2bf(acc[j]);
            }
        }
    }
}

// ---------------------------------------------------------------------------
// Pack layer weights (filt tap0/1, gate tap0/1, res) into exact WMMA-B lane
// layout: wpack[(((mat*4+otile)*2+ks)*32+lane)*8 + j] = bf16x2 of w[o][k0],w[o][k0+1]
// ---------------------------------------------------------------------------
__global__ void k_prepw(const float* __restrict__ fw, const float* __restrict__ gw,
                        const float* __restrict__ rw, u32* __restrict__ wpack, int layer) {
    const int idx = blockIdx.x * 256 + threadIdx.x; // 5*4*2*32*8 = 10240
    if (idx >= 10240) return;
    const int j = idx & 7;
    const int lane = (idx >> 3) & 31;
    const int ks = (idx >> 8) & 1;
    const int otile = (idx >> 9) & 3;
    const int mat = idx >> 11;
    const int o = otile * 16 + (lane & 15);
    const int k0 = ks * 32 + (j < 4 ? (lane >> 4) * 8 + 2 * j : 16 + (lane >> 4) * 8 + 2 * (j - 4));
    float v0, v1;
    if (mat < 4) {
        const float* w = (mat < 2 ? fw : gw) + (size_t)layer * HH * HH * 2;
        const int tap = mat & 1;
        v0 = w[(o * HH + k0) * 2 + tap];
        v1 = w[(o * HH + k0 + 1) * 2 + tap];
    } else {
        const float* w = rw + (size_t)layer * HH * HH;
        v0 = w[o * HH + k0];
        v1 = w[o * HH + k0 + 1];
    }
    wpack[idx] = pack2(v0, v1);
}

// ---------------------------------------------------------------------------
// Fused dilated-conv layer. Rows r = ((b*NN+m)*65 + t), t in [0,64].
//   f = tanh(xs[t-d]·Wf0 + xs[t]·Wf1 + fb)   (taps zero-padded via row guards)
//   g = sigmoid(... gate ...)
//   xt = BN(f*g);  t<64: h' = xt + (xs[t]·Wr + rb) -> hbf ; t==64: skip += xt
// 10 chained WMMA per wave (16 rows x 16 output channels).
// ---------------------------------------------------------------------------
__global__ void k_conv(const u16* __restrict__ xs, const u32* __restrict__ wpack,
                       const float* __restrict__ fb, const float* __restrict__ gb,
                       const float* __restrict__ rb, const float* __restrict__ bng,
                       const float* __restrict__ bnb, const float* __restrict__ bnm,
                       const float* __restrict__ bnv, u16* __restrict__ hbf,
                       float* __restrict__ skip, int d) {
    const int gw = blockIdx.x * 4 + (threadIdx.x >> 5);
    const int rowTile = gw >> 2;  // 33280 row tiles
    const int otile = gw & 3;     // 4 channel tiles
    const int lane = threadIdx.x & 31;
    const int mrow = lane & 15;
    const int khalf = lane >> 4;

    const int r = rowTile * 16 + mrow;
    const int q = r / TOUT;
    const int t = r - q * TOUT;
    const int m = q & (NN - 1);
    const int b = q >> 9;
    const bool tap0_ok = (t >= d);
    const bool tap1_ok = (t < TT);

    const u16* base = xs + ((size_t)(b * NN + m)) * TT * HH;
    const u16* a0p = base + (t - d) * HH;
    const u16* a1p = base + t * HH;

    v8f aF = {0.f,0.f,0.f,0.f,0.f,0.f,0.f,0.f};
    v8f aG = {0.f,0.f,0.f,0.f,0.f,0.f,0.f,0.f};
    v8f aR = {0.f,0.f,0.f,0.f,0.f,0.f,0.f,0.f};

    #pragma unroll
    for (int ks = 0; ks < 2; ++ks) {
        BF16x16 A0, A1, Bf0, Bf1, Bg0, Bg1, Br;
        #pragma unroll
        for (int j = 0; j < 8; ++j) {
            const int ofs = ks * 32 + (j < 4 ? khalf * 8 + 2 * j
                                             : 16 + khalf * 8 + 2 * (j - 4));
            A0.u[j] = tap0_ok ? *(const u32*)(a0p + ofs) : 0u;
            A1.u[j] = tap1_ok ? *(const u32*)(a1p + ofs) : 0u;
        }
        const u32* w0 = wpack + (((0 * 4 + otile) * 2 + ks) * 32 + lane) * 8;
        const u32* w1 = wpack + (((1 * 4 + otile) * 2 + ks) * 32 + lane) * 8;
        const u32* w2 = wpack + (((2 * 4 + otile) * 2 + ks) * 32 + lane) * 8;
        const u32* w3 = wpack + (((3 * 4 + otile) * 2 + ks) * 32 + lane) * 8;
        const u32* w4 = wpack + (((4 * 4 + otile) * 2 + ks) * 32 + lane) * 8;
        #pragma unroll
        for (int j = 0; j < 8; ++j) {
            Bf0.u[j] = w0[j]; Bf1.u[j] = w1[j];
            Bg0.u[j] = w2[j]; Bg1.u[j] = w3[j];
            Br.u[j]  = w4[j];
        }
        aF = wmma_bf16(A0, Bf0, aF);
        aF = wmma_bf16(A1, Bf1, aF);
        aG = wmma_bf16(A0, Bg0, aG);
        aG = wmma_bf16(A1, Bg1, aG);
        aR = wmma_bf16(A1, Br, aR);
    }

    const int o = otile * 16 + (lane & 15);
    const float fbias = fb[o], gbias = gb[o], rbias = rb[o];
    const float sc = bng[o] * rsqrtf(bnv[o] + 1e-5f);
    const float sh = bnb[o] - bnm[o] * sc;

    #pragma unroll
    for (int j = 0; j < 8; ++j) {
        const int r2 = rowTile * 16 + j + 8 * khalf;
        const int q2 = r2 / TOUT;
        const int t2 = r2 - q2 * TOUT;
        const int m2 = q2 & (NN - 1);
        const int b2 = q2 >> 9;
        const float f = tanhf(aF[j] + fbias);
        const float g = 1.0f / (1.0f + __expf(-(aG[j] + gbias)));
        const float xt = (f * g) * sc + sh;
        if (t2 < TT) {
            const float hv = xt + aR[j] + rbias;
            hbf[(((size_t)(b2 * HH + o)) * TT + t2) * NN + m2] = f2bf(hv);
        } else { // t2 == 64: only slice of skip that reaches the output head
            skip[(size_t)(b2 * HH + o) * NN + m2] += xt;
        }
    }
}

// ---------------------------------------------------------------------------
// Head: v[b][m] = end2( relu(end1( relu(skip[b][:,m]) )) )
// ---------------------------------------------------------------------------
__global__ void k_end1(const float* __restrict__ skip, const float* __restrict__ e1w,
                       const float* __restrict__ e1b, const float* __restrict__ e2w,
                       const float* __restrict__ e2b, float* __restrict__ vbuf) {
    const int idx = blockIdx.x * 256 + threadIdx.x; // BB*NN = 8192
    const int b = idx >> 9, m = idx & (NN - 1);
    float s[HH];
    #pragma unroll
    for (int o = 0; o < HH; ++o)
        s[o] = fmaxf(skip[(size_t)(b * HH + o) * NN + m], 0.f);
    float acc = e2b[0];
    for (int o2 = 0; o2 < HH; ++o2) {
        float u = e1b[o2];
        #pragma unroll
        for (int o = 0; o < HH; ++o) u += s[o] * e1w[o2 * HH + o];
        acc += fmaxf(u, 0.f) * e2w[o2];
    }
    vbuf[idx] = acc;
}

// out[b][mm] = sum_m v[b][m] * bias_w[mm][m] + bias_b[mm]
__global__ void k_end2(const float* __restrict__ vbuf, const float* __restrict__ bw,
                       const float* __restrict__ bbias, float* __restrict__ out) {
    const int idx = blockIdx.x * 256 + threadIdx.x; // 8192
    const int b = idx >> 9, mm = idx & (NN - 1);
    float acc = bbias[mm];
    for (int m = 0; m < NN; ++m) acc += vbuf[b * NN + m] * bw[mm * NN + m];
    out[idx] = acc;
}

extern "C" void kernel_launch(void* const* d_in, const int* in_sizes, int n_in,
                              void* d_out, int out_size, void* d_ws, size_t ws_size,
                              hipStream_t stream) {
    const float* x       = (const float*)d_in[0];
    const float* sadj    = (const float*)d_in[1];
    const float* e1      = (const float*)d_in[2];
    const float* e2      = (const float*)d_in[3];
    const float* start_w = (const float*)d_in[4];
    const float* start_b = (const float*)d_in[5];
    const float* filt_w  = (const float*)d_in[6];
    const float* filt_b  = (const float*)d_in[7];
    const float* gate_w  = (const float*)d_in[8];
    const float* gate_b  = (const float*)d_in[9];
    const float* res_w   = (const float*)d_in[10];
    const float* res_b   = (const float*)d_in[11];
    const float* bn_g    = (const float*)d_in[12];
    const float* bn_b    = (const float*)d_in[13];
    const float* bn_m    = (const float*)d_in[14];
    const float* bn_v    = (const float*)d_in[15];
    const float* end1_w  = (const float*)d_in[16];
    const float* end1_b  = (const float*)d_in[17];
    const float* end2_w  = (const float*)d_in[18];
    const float* end2_b  = (const float*)d_in[19];
    const float* bias_w  = (const float*)d_in[20];
    const float* bias_b  = (const float*)d_in[21];
    float* out = (float*)d_out;

    char* ws = (char*)d_ws;
    const size_t SZ_ADJT = (size_t)NN * NN * 2;                  // 0.5 MB bf16
    const size_t SZ_H    = (size_t)BB * HH * TT * NN * 2;        // 67 MB bf16
    const size_t SZ_XS   = (size_t)BB * NN * TT * HH * 2;        // 67 MB bf16
    const size_t SZ_WPK  = 10240 * 4;
    const size_t SZ_SKIP = (size_t)BB * HH * NN * 4;             // 2 MB f32
    u16* adjT  = (u16*)(ws);
    u16* hbf   = (u16*)(ws + SZ_ADJT);
    u16* xs    = (u16*)(ws + SZ_ADJT + SZ_H);
    u32* wpack = (u32*)(ws + SZ_ADJT + SZ_H + SZ_XS);
    float* skip= (float*)(ws + SZ_ADJT + SZ_H + SZ_XS + SZ_WPK);
    float* vbuf= (float*)(ws + SZ_ADJT + SZ_H + SZ_XS + SZ_WPK + SZ_SKIP);

    k_adj<<<NN, 256, 0, stream>>>(e1, e2, sadj, adjT);
    k_start<<<(BB * HH * TT * NN) / 256, 256, 0, stream>>>(x, start_w, start_b, hbf);
    k_zero<<<(BB * HH * NN) / 256, 256, 0, stream>>>(skip, BB * HH * NN);

    for (int i = 0; i < 4; ++i) {
        const int d = 1 << i;
        // diffusion: 2048 macro row tiles * 16 macro col tiles, 4 waves/block
        k_diffuse<<<(2048 * 16) / 4, 128, 0, stream>>>(hbf, adjT, xs);
        k_prepw<<<40, 256, 0, stream>>>(filt_w, gate_w, res_w, wpack, i);
        // conv: 33280 row tiles * 4 otile, 4 waves/block
        k_conv<<<33280, 128, 0, stream>>>(xs, wpack,
                                          filt_b + i * HH, gate_b + i * HH, res_b + i * HH,
                                          bn_g + i * HH, bn_b + i * HH, bn_m + i * HH, bn_v + i * HH,
                                          hbf, skip, d);
    }

    k_end1<<<(BB * NN) / 256, 256, 0, stream>>>(skip, end1_w, end1_b, end2_w, end2_b, vbuf);
    k_end2<<<(BB * NN) / 256, 256, 0, stream>>>(vbuf, bias_w, bias_b, out);
}
